// RADFA_80479097193022
// MI455X (gfx1250) — matrix operand order repaired and checked by reading, not compiled, not run on hardware
//
#include <hip/hip_runtime.h>
#include <math.h>

// ---------------------------------------------------------------------------
// Problem constants (from reference)
// ---------------------------------------------------------------------------
#define BB   2
#define SS   2048
#define DD   256
#define QKD  256
#define HH   8
#define HDD  32
#define MLPD 1024
#define TOPKN 8
#define RR   2048
#define MDD  64
#define SCALE_QK 0.17677669529663687f  // 1/sqrt(32)

// GEMM block tiling
#define BM 128
#define BN 128
#define BK 32

typedef __bf16 bf16;
typedef __attribute__((ext_vector_type(16))) __bf16 v16bf;
typedef __attribute__((ext_vector_type(8)))  float  v8f;
typedef __attribute__((ext_vector_type(4)))  unsigned int u32x4;
typedef __attribute__((ext_vector_type(8)))  unsigned int u32x8;

#define WMMA_BF16(a, b, c) \
  __builtin_amdgcn_wmma_f32_16x16x32_bf16(false, (a), false, (b), (short)0, (c), false, false)

// ---------------------------------------------------------------------------
// wave32 WMMA fragment layouts (CDNA5 ISA §7.12.2)
// A (16x32 bf16): lane holds row m=lane&15.  lanes 0-15: elems 0..7 -> K 0..7,
//   elems 8..15 -> K 16..23; lanes 16-31: K 8..15 / 24..31.
// B (32x16 bf16): lane holds col n=lane&15.  lanes 0-15: elems 0..7 -> K 0..7,
//   elems 8..15 -> K 16..23; lanes 16-31: K 8..15 / 24..31 (same as A).
// C/D (16x16 f32): lane holds col n=lane&15; vgpr v -> row m = v + 8*(lane>=16).
// ---------------------------------------------------------------------------
__device__ __forceinline__ int a_frag_k(int lane, int e) {
  return ((e >> 3) << 4) + ((lane & 16) ? 8 : 0) + (e & 7);
}
__device__ __forceinline__ int b_frag_k(int lane, int e) {
  return ((e >> 3) << 4) + ((lane & 16) ? 8 : 0) + (e & 7);
}

// ---------------------------------------------------------------------------
// 1) Region cosine-sim + top-8 + adjacency  ->  ext[R, S] uint8 mask table
// ---------------------------------------------------------------------------
__global__ void build_mask_kernel(const float* __restrict__ rm,
                                  const unsigned char* __restrict__ adj,
                                  unsigned char* __restrict__ ext) {
  __shared__ float nmr[MDD];
  __shared__ float simbuf[RR];
  __shared__ float redv[256];
  __shared__ int   redi[256];
  __shared__ float norm_s;
  const int r = blockIdx.x;
  const int t = threadIdx.x;

  if (t < MDD) nmr[t] = rm[(size_t)r * MDD + t];
  __syncthreads();
  if (t == 0) {
    float s = 0.f;
    for (int d = 0; d < MDD; ++d) s += nmr[d] * nmr[d];
    norm_s = sqrtf(s) + 1e-8f;
  }
  __syncthreads();
  const float inr = 1.0f / norm_s;

  for (int c = t; c < RR; c += 256) {
    float dot = 0.f, sq = 0.f;
    const float* rc = rm + (size_t)c * MDD;
    for (int d = 0; d < MDD; ++d) { float w = rc[d]; dot += nmr[d] * w; sq += w * w; }
    float sim = dot * inr / (sqrtf(sq) + 1e-8f);
    simbuf[c] = (c == r) ? -INFINITY : sim;
    ext[(size_t)r * SS + c] = adj[(size_t)r * RR + c] ? 1 : 0;
  }
  __syncthreads();

  for (int it = 0; it < TOPKN; ++it) {
    float bv = -INFINITY; int bi = 0x7fffffff;
    for (int c = t; c < RR; c += 256) {
      float v = simbuf[c];
      if (v > bv || (v == bv && c < bi)) { bv = v; bi = c; }
    }
    redv[t] = bv; redi[t] = bi;
    __syncthreads();
    for (int off = 128; off > 0; off >>= 1) {
      if (t < off) {
        float v2 = redv[t + off]; int i2 = redi[t + off];
        if (v2 > redv[t] || (v2 == redv[t] && i2 < redi[t])) { redv[t] = v2; redi[t] = i2; }
      }
      __syncthreads();
    }
    if (t == 0) {
      int bidx = redi[0];
      ext[(size_t)r * SS + bidx] = 1;
      simbuf[bidx] = -INFINITY;
    }
    __syncthreads();
  }
}

// ---------------------------------------------------------------------------
// 2) LayerNorm (D=256): one block per row -> bf16 out
// ---------------------------------------------------------------------------
__global__ void layernorm_kernel(const float* __restrict__ x,
                                 const float* __restrict__ g,
                                 const float* __restrict__ bvec,
                                 bf16* __restrict__ out) {
  __shared__ float red[256];
  const int row = blockIdx.x, t = threadIdx.x;
  const float v = x[(size_t)row * DD + t];
  red[t] = v; __syncthreads();
  for (int off = 128; off > 0; off >>= 1) { if (t < off) red[t] += red[t + off]; __syncthreads(); }
  const float mean = red[0] * (1.0f / DD);
  __syncthreads();
  const float d = v - mean;
  red[t] = d * d; __syncthreads();
  for (int off = 128; off > 0; off >>= 1) { if (t < off) red[t] += red[t + off]; __syncthreads(); }
  const float var = red[0] * (1.0f / DD);
  out[(size_t)row * DD + t] = (bf16)(d * rsqrtf(var + 1e-5f) * g[t] + bvec[t]);
}

// ---------------------------------------------------------------------------
// 3) Weight convert + transpose: w[K,N] f32 -> wt[N,K] bf16
// ---------------------------------------------------------------------------
__global__ void wconvT_kernel(const float* __restrict__ w, bf16* __restrict__ wt,
                              int K, int N) {
  int i = blockIdx.x * 256 + threadIdx.x;
  if (i < K * N) {
    int k = i / N, n = i % N;
    wt[(size_t)n * K + k] = (bf16)w[i];
  }
}

// ---------------------------------------------------------------------------
// 4) WMMA GEMM with LDS staging:
//      A tile (128x32) staged by the Tensor Data Mover (tensor_load_to_lds),
//      B tile (128x32) staged by per-thread global_load_async_to_lds_b128.
//    Block = 256 threads (8 waves) computes a 128x128 output tile; each wave
//    owns a 128x16 strip (8 accumulators -> 8 WMMAs per K-step).
//    Dynamic LDS so the tile base offsets are literal wave-LDS addresses:
//      ldsA @ 0 (8KB), ldsB @ 8192 (8KB).
//    ACT: 0 = none, 1 = exact GELU.
// ---------------------------------------------------------------------------
template <int ACT>
__global__ void gemm_kernel(const bf16* __restrict__ A, const bf16* __restrict__ Bt,
                            const float* __restrict__ bias,
                            float* __restrict__ Cf, bf16* __restrict__ Cb,
                            int M, int N, int K) {
  extern __shared__ char smem_raw[];
  bf16* ldsA = (bf16*)smem_raw;              // wave-LDS offset 0
  bf16* ldsB = (bf16*)(smem_raw + 8192);     // wave-LDS offset 8192
  const int tid  = threadIdx.x;
  const int lane = tid & 31;
  const int wave = tid >> 5;
  const int m0 = blockIdx.y * BM;
  const int n0 = blockIdx.x * BN;

  v8f acc[8];
#pragma unroll
  for (int i = 0; i < 8; ++i) acc[i] = (v8f){};

  for (int k0 = 0; k0 < K; k0 += BK) {
    // ---- A tile: TDM DMA, issued by wave 0 only (EXEC ignored per-wave) ----
    if (tid < 32) {
      unsigned long long ga =
          (unsigned long long)(uintptr_t)(A + (size_t)m0 * K + k0);
      unsigned galo = (unsigned)__builtin_amdgcn_readfirstlane((int)(ga & 0xffffffffu));
      unsigned gahi = (unsigned)__builtin_amdgcn_readfirstlane((int)(ga >> 32));
      // D# group 0: count=1 | lds_addr | global_addr | type=2 (CDNA5 ISA)
      u32x4 g0;
      g0[0] = 1u;
      g0[1] = 0u;                                   // lds_addr = 0 (ldsA)
      g0[2] = galo;
      g0[3] = (gahi & 0x01ffffffu) | (2u << 30);
      // D# group 1: mask=0 | data_size=2B | dims/tile/stride (§8.4)
      u32x8 g1;
      g1[0] = 1u << 16;                             // data_size=1 -> 2 bytes
      g1[1] = ((unsigned)BK & 0xffffu) << 16;       // tensor_dim0 lo16 = 32
      g1[2] = ((unsigned)BM & 0xffffu) << 16;       // tensor_dim0 hi=0 | tensor_dim1 lo16 = 128
      g1[3] = ((unsigned)BK & 0xffffu) << 16;       // tensor_dim1 hi=0 | tile_dim0 = 32
      g1[4] = (unsigned)BM & 0xffffu;               // tile_dim1 = 128, tile_dim2 = 0
      g1[5] = (unsigned)K;                          // tensor_dim0_stride lo32 (elements)
      g1[6] = 0u;
      g1[7] = 0u;
      asm volatile("tensor_load_to_lds %0, %1" :: "s"(g0), "s"(g1) : "memory");
      __builtin_amdgcn_s_wait_tensorcnt(0);
    }
    // ---- B tile: async global->LDS, 2 x 16B chunks per thread (8KB total) ----
    {
      const int c0 = tid;            // chunk ids (16B each); 4 chunks per 64B row
      const int c1 = tid + 256;
      const bf16* s0 = Bt + (size_t)(n0 + (c0 >> 2)) * K + k0 + (c0 & 3) * 8;
      const bf16* s1 = Bt + (size_t)(n0 + (c1 >> 2)) * K + k0 + (c1 & 3) * 8;
      unsigned d0 = 8192u + (unsigned)c0 * 16u;
      unsigned d1 = 8192u + (unsigned)c1 * 16u;
      asm volatile("global_load_async_to_lds_b128 %0, %1, off"
                   :: "v"(d0), "v"(s0) : "memory");
      asm volatile("global_load_async_to_lds_b128 %0, %1, off"
                   :: "v"(d1), "v"(s1) : "memory");
      asm volatile("s_wait_asynccnt 0x0" ::: "memory");
    }
    __syncthreads();

    // ---- compute: one B fragment per wave, 8 m-subtiles ----
    v16bf bfrag;
    const int nloc = wave * 16 + (lane & 15);
#pragma unroll
    for (int e = 0; e < 16; ++e) bfrag[e] = ldsB[nloc * BK + b_frag_k(lane, e)];
#pragma unroll
    for (int mt = 0; mt < 8; ++mt) {
      v16bf afrag;
      const int mloc = mt * 16 + (lane & 15);
#pragma unroll
      for (int e = 0; e < 16; ++e) afrag[e] = ldsA[mloc * BK + a_frag_k(lane, e)];
      acc[mt] = WMMA_BF16(afrag, bfrag, acc[mt]);
    }
    __syncthreads();   // protect LDS tiles before next staging
  }

  // ---- epilogue ----
  const int ncol = n0 + wave * 16 + (lane & 15);
  const int moff = (lane & 16) ? 8 : 0;
  const float bb = bias[ncol];
#pragma unroll
  for (int mt = 0; mt < 8; ++mt) {
#pragma unroll
    for (int vv = 0; vv < 8; ++vv) {
      const int m = m0 + mt * 16 + moff + vv;
      float val = acc[mt][vv] + bb;
      if (ACT == 1) val = 0.5f * val * (1.0f + erff(val * 0.70710678118654752f));
      const size_t off = (size_t)m * N + ncol;
      if (Cf) Cf[off] = val;
      if (Cb) Cb[off] = (bf16)val;
    }
  }
}

// ---------------------------------------------------------------------------
// 5) Fused region attention, flash-style; one wave per (b, h, 16-query tile).
//    Faithful semantics: non-neighbor scores are 0 (NOT -inf), softmax over
//    ALL j  =>  exp(0)=1 terms included in numerator and denominator.
// ---------------------------------------------------------------------------
__global__ void attn_kernel(const bf16* __restrict__ q, const bf16* __restrict__ k,
                            const bf16* __restrict__ v, const int* __restrict__ reg,
                            const unsigned char* __restrict__ ext,
                            bf16* __restrict__ ctx) {
  __shared__ float pexp[16][32];
  __shared__ float denom_s[16];
  __shared__ int   regs[16];
  const int lane = threadIdx.x;
  const int i0 = blockIdx.x * 16;
  const int h  = blockIdx.y;
  const int b  = blockIdx.z;
  const int col0 = h * HDD;
  const int moff = (lane & 16) ? 8 : 0;

  if (lane < 16) regs[lane] = reg[i0 + lane];
  __syncthreads();

  const bf16* qrow = q + ((size_t)(b * SS + i0 + (lane & 15))) * QKD + col0;
  v16bf aq;
#pragma unroll
  for (int e = 0; e < 16; ++e) aq[e] = qrow[a_frag_k(lane, e)];

  v8f acc0 = {}, acc1 = {};
  float denom = 0.f;

  for (int j0 = 0; j0 < SS; j0 += 32) {
#pragma unroll
    for (int half = 0; half < 2; ++half) {
      const int jb = j0 + half * 16;
      const bf16* krow = k + ((size_t)(b * SS + jb + (lane & 15))) * QKD + col0;
      v16bf bk;
#pragma unroll
      for (int e = 0; e < 16; ++e) bk[e] = krow[b_frag_k(lane, e)];
      v8f c = {};
      c = WMMA_BF16(aq, bk, c);
      const int n = lane & 15;
      const int j = jb + n;
#pragma unroll
      for (int vv = 0; vv < 8; ++vv) {
        const int m = moff + vv;
        const float s  = c[vv] * SCALE_QK;
        const float sc = ext[(size_t)regs[m] * SS + j] ? s : 0.0f;
        pexp[m][half * 16 + n] = expf(sc);
      }
    }
    __syncthreads();

    if (lane < 16) {
      float t = 0.f;
#pragma unroll
      for (int jj = 0; jj < 32; ++jj) t += pexp[lane][jj];
      denom += t;
    }

    v16bf ap;
#pragma unroll
    for (int e = 0; e < 16; ++e) ap[e] = (bf16)pexp[lane & 15][a_frag_k(lane, e)];

    v16bf bv0, bv1;
#pragma unroll
    for (int e = 0; e < 16; ++e) {
      const int j = j0 + b_frag_k(lane, e);
      const bf16* vrow = v + ((size_t)(b * SS + j)) * QKD + col0 + (lane & 15);
      bv0[e] = vrow[0];
      bv1[e] = vrow[16];
    }
    acc0 = WMMA_BF16(ap, bv0, acc0);
    acc1 = WMMA_BF16(ap, bv1, acc1);
    __syncthreads();
  }

  if (lane < 16) denom_s[lane] = denom;
  __syncthreads();

  const int n = lane & 15;
#pragma unroll
  for (int vv = 0; vv < 8; ++vv) {
    const int m = moff + vv;
    const float dinv = 1.0f / denom_s[m];
    const size_t o = ((size_t)(b * SS + i0 + m)) * QKD + col0;
    ctx[o + n]      = (bf16)(acc0[vv] * dinv);
    ctx[o + 16 + n] = (bf16)(acc1[vv] * dinv);
  }
}

// ---------------------------------------------------------------------------
// 6) Elementwise helpers
// ---------------------------------------------------------------------------
__global__ void concat_kernel(const float* __restrict__ x, const float* __restrict__ ao,
                              bf16* __restrict__ cat) {
  const int i = blockIdx.x * 256 + threadIdx.x;
  const int row = i / DD, cpos = i % DD;
  cat[(size_t)row * (2 * DD) + cpos]      = (bf16)x[i];
  cat[(size_t)row * (2 * DD) + DD + cpos] = (bf16)ao[i];
}

__global__ void gate_fuse_kernel(const float* __restrict__ gl, const float* __restrict__ x,
                                 const float* __restrict__ ao, float* __restrict__ fused) {
  const int i = blockIdx.x * 256 + threadIdx.x;
  const float g = 1.0f / (1.0f + expf(-gl[i]));
  fused[i] = g * x[i] + (1.0f - g) * ao[i];
}

__global__ void add_kernel(const float* __restrict__ a, const float* __restrict__ b,
                           float* __restrict__ o) {
  const int i = blockIdx.x * 256 + threadIdx.x;
  o[i] = a[i] + b[i];
}

// ---------------------------------------------------------------------------
// Launcher
// ---------------------------------------------------------------------------
extern "C" void kernel_launch(void* const* d_in, const int* in_sizes, int n_in,
                              void* d_out, int out_size, void* d_ws, size_t ws_size,
                              hipStream_t stream) {
  (void)in_sizes; (void)n_in; (void)out_size; (void)ws_size;
  const float* x    = (const float*)d_in[0];
  const int*   reg  = (const int*)d_in[1];
  const unsigned char* adj = (const unsigned char*)d_in[2];  // bool array: 1 byte/elem
  const float* rm   = (const float*)d_in[3];
  const float* wq   = (const float*)d_in[4];  const float* bq  = (const float*)d_in[5];
  const float* wk   = (const float*)d_in[6];  const float* bk  = (const float*)d_in[7];
  const float* wv   = (const float*)d_in[8];  const float* bvv = (const float*)d_in[9];
  const float* wo   = (const float*)d_in[10]; const float* bo  = (const float*)d_in[11];
  const float* ln1g = (const float*)d_in[12]; const float* ln1b = (const float*)d_in[13];
  const float* ln2g = (const float*)d_in[14]; const float* ln2b = (const float*)d_in[15];
  const float* gw   = (const float*)d_in[16]; const float* gb  = (const float*)d_in[17];
  const float* f1w  = (const float*)d_in[18]; const float* f1b = (const float*)d_in[19];
  const float* f2w  = (const float*)d_in[20]; const float* f2b = (const float*)d_in[21];
  float* out = (float*)d_out;

  const int Mrows = BB * SS;  // 4096
  char* p = (char*)d_ws;
  auto carve = [&](size_t bytes) -> char* {
    char* r = p; p += (bytes + 255) & ~(size_t)255; return r;
  };
  unsigned char* ext = (unsigned char*)carve((size_t)RR * SS);
  bf16*  xn       = (bf16*) carve((size_t)Mrows * DD * 2);
  bf16*  qb       = (bf16*) carve((size_t)Mrows * QKD * 2);
  bf16*  kb       = (bf16*) carve((size_t)Mrows * QKD * 2);
  bf16*  vb       = (bf16*) carve((size_t)Mrows * QKD * 2);
  bf16*  ctx      = (bf16*) carve((size_t)Mrows * QKD * 2);
  float* attn_out = (float*)carve((size_t)Mrows * DD * 4);
  bf16*  cat      = (bf16*) carve((size_t)Mrows * 2 * DD * 2);
  float* gate_lin = (float*)carve((size_t)Mrows * DD * 4);
  float* fused    = (float*)carve((size_t)Mrows * DD * 4);
  bf16*  hb       = (bf16*) carve((size_t)Mrows * DD * 2);
  bf16*  mid      = (bf16*) carve((size_t)Mrows * MLPD * 2);
  float* ffn_out  = (float*)carve((size_t)Mrows * DD * 4);
  bf16*  wqT  = (bf16*)carve((size_t)DD * QKD * 2);
  bf16*  wkT  = (bf16*)carve((size_t)DD * QKD * 2);
  bf16*  wvT  = (bf16*)carve((size_t)DD * QKD * 2);
  bf16*  woT  = (bf16*)carve((size_t)QKD * DD * 2);
  bf16*  gwT  = (bf16*)carve((size_t)2 * DD * DD * 2);
  bf16*  f1T  = (bf16*)carve((size_t)DD * MLPD * 2);
  bf16*  f2T  = (bf16*)carve((size_t)MLPD * DD * 2);

  const size_t GEMM_LDS = 16384;  // ldsA 8KB + ldsB 8KB

  build_mask_kernel<<<RR, 256, 0, stream>>>(rm, adj, ext);

  wconvT_kernel<<<(DD * QKD + 255) / 256, 256, 0, stream>>>(wq, wqT, DD, QKD);
  wconvT_kernel<<<(DD * QKD + 255) / 256, 256, 0, stream>>>(wk, wkT, DD, QKD);
  wconvT_kernel<<<(DD * QKD + 255) / 256, 256, 0, stream>>>(wv, wvT, DD, QKD);
  wconvT_kernel<<<(QKD * DD + 255) / 256, 256, 0, stream>>>(wo, woT, QKD, DD);
  wconvT_kernel<<<(2 * DD * DD + 255) / 256, 256, 0, stream>>>(gw, gwT, 2 * DD, DD);
  wconvT_kernel<<<(DD * MLPD + 255) / 256, 256, 0, stream>>>(f1w, f1T, DD, MLPD);
  wconvT_kernel<<<(MLPD * DD + 255) / 256, 256, 0, stream>>>(f2w, f2T, MLPD, DD);

  layernorm_kernel<<<Mrows, 256, 0, stream>>>(x, ln1g, ln1b, xn);
  dim3 gqkv(QKD / BN, Mrows / BM);   // (2, 32)
  gemm_kernel<0><<<gqkv, 256, GEMM_LDS, stream>>>(xn, wqT, bq,  nullptr, qb, Mrows, QKD, DD);
  gemm_kernel<0><<<gqkv, 256, GEMM_LDS, stream>>>(xn, wkT, bk,  nullptr, kb, Mrows, QKD, DD);
  gemm_kernel<0><<<gqkv, 256, GEMM_LDS, stream>>>(xn, wvT, bvv, nullptr, vb, Mrows, QKD, DD);

  attn_kernel<<<dim3(SS / 16, HH, BB), 32, 0, stream>>>(qb, kb, vb, reg, ext, ctx);
  gemm_kernel<0><<<dim3(DD / BN, Mrows / BM), 256, GEMM_LDS, stream>>>(
      ctx, woT, bo, attn_out, nullptr, Mrows, DD, QKD);

  concat_kernel<<<Mrows * DD / 256, 256, 0, stream>>>(x, attn_out, cat);
  gemm_kernel<0><<<dim3(DD / BN, Mrows / BM), 256, GEMM_LDS, stream>>>(
      cat, gwT, gb, gate_lin, nullptr, Mrows, DD, 2 * DD);
  gate_fuse_kernel<<<Mrows * DD / 256, 256, 0, stream>>>(gate_lin, x, attn_out, fused);

  layernorm_kernel<<<Mrows, 256, 0, stream>>>(fused, ln2g, ln2b, hb);
  gemm_kernel<1><<<dim3(MLPD / BN, Mrows / BM), 256, GEMM_LDS, stream>>>(
      hb, f1T, f1b, nullptr, mid, Mrows, MLPD, DD);
  gemm_kernel<0><<<dim3(DD / BN, Mrows / BM), 256, GEMM_LDS, stream>>>(
      mid, f2T, f2b, ffn_out, nullptr, Mrows, DD, MLPD);
  add_kernel<<<Mrows * DD / 256, 256, 0, stream>>>(fused, ffn_out, out);
}
